// AIGStateEncoder_7035156430972
// MI455X (gfx1250) — compile-verified, matrix operand order checked by reading, not run on hardware
//
#include <hip/hip_runtime.h>
#include <hip/hip_bf16.h>

// ---------------------------------------------------------------------------
// CDNA5 (gfx1250) wave32 WMMA types
// ---------------------------------------------------------------------------
typedef __attribute__((ext_vector_type(16))) __bf16 v16bf;
typedef __attribute__((ext_vector_type(8)))  float  v8f;

static __device__ __forceinline__ v8f wmma_bf16(v16bf a, v16bf b, v8f c) {
    // D = A(16x32 bf16) * B(32x16 bf16) + C(16x16 f32)
    return __builtin_amdgcn_wmma_f32_16x16x32_bf16(
        false, a, false, b, (short)0, c, false, false);
}

// A-matrix (16x32, 16-bit) layout: lane L holds row M=L&15.
//  lanes 0-15:  elem i -> K = (i<8 ? i : i+8)
//  lanes 16-31: elem i -> K = (i<8 ? i : i+8) + 8
// tile points at element (row0, k0); ld = floats between rows.
static __device__ __forceinline__ v16bf load_A_f32(const float* tile, int ld) {
    int lane = threadIdx.x & 31;
    int row  = lane & 15;
    int hi   = lane >> 4;
    const float* r = tile + (size_t)row * ld + hi * 8;
    float4 x0 = *(const float4*)(r + 0);
    float4 x1 = *(const float4*)(r + 4);
    float4 x2 = *(const float4*)(r + 16);
    float4 x3 = *(const float4*)(r + 20);
    v16bf a;
    a[0]=(__bf16)x0.x; a[1]=(__bf16)x0.y; a[2]=(__bf16)x0.z; a[3]=(__bf16)x0.w;
    a[4]=(__bf16)x1.x; a[5]=(__bf16)x1.y; a[6]=(__bf16)x1.z; a[7]=(__bf16)x1.w;
    a[8]=(__bf16)x2.x; a[9]=(__bf16)x2.y; a[10]=(__bf16)x2.z; a[11]=(__bf16)x2.w;
    a[12]=(__bf16)x3.x; a[13]=(__bf16)x3.y; a[14]=(__bf16)x3.z; a[15]=(__bf16)x3.w;
    return a;
}

// Same layout, bf16 source (row-major, ld elements between rows).
static __device__ __forceinline__ v16bf load_A_bf16(const __bf16* tile, int ld) {
    int lane = threadIdx.x & 31;
    int row  = lane & 15;
    int hi   = lane >> 4;
    const __bf16* r = tile + (size_t)row * ld + hi * 8;
    v16bf a;
#pragma unroll
    for (int i = 0; i < 8; ++i) a[i] = r[i];
#pragma unroll
    for (int i = 0; i < 8; ++i) a[8 + i] = r[16 + i];
    return a;
}

// B-matrix (32x16, 16-bit), source row-major [K x N]: lane L holds column N=L&15,
// K follows the same split as A. tile points at element (k0, col0); ld = row stride.
// Across lanes 0-15 a given element index reads 16 consecutive floats (coalesced).
static __device__ __forceinline__ v16bf load_B_kn(const float* tile, int ld) {
    int lane = threadIdx.x & 31;
    int col  = lane & 15;
    int hi   = lane >> 4;
    const float* p = tile + col + (size_t)(hi * 8) * ld;
    v16bf b;
#pragma unroll
    for (int i = 0; i < 8; ++i) b[i] = (__bf16)p[(size_t)i * ld];
    p += (size_t)16 * ld;
#pragma unroll
    for (int i = 0; i < 8; ++i) b[8 + i] = (__bf16)p[(size_t)i * ld];
    return b;
}

static __device__ __forceinline__ float sigmf(float x) { return 1.f / (1.f + __expf(-x)); }

// order-preserving float<->uint for atomicMax on floats
static __device__ __forceinline__ unsigned enc_f(float x) {
    unsigned u = __float_as_uint(x);
    return (u & 0x80000000u) ? ~u : (u | 0x80000000u);
}
static __device__ __forceinline__ float dec_f(unsigned k) {
    unsigned u = (k & 0x80000000u) ? (k ^ 0x80000000u) : ~k;
    return __uint_as_float(u);
}
#define ENC_NEG_INF 0x007FFFFFu  // enc(-inf)

// ---------------------------------------------------------------------------
// Generic tiled GEMM:  C = act( A @ B + bias , addend )
//   A: [M,K] f32 row-major (lda).  B: [K,N] f32 row-major (ldb).
//   bias: [N] or null. addend: addend[m*ldadd+n] or null (ldadd=0 broadcasts a row).
//   relu: 0 = none
//         1 = relu BEFORE addend-add   ("x + relu(f(x))" residual pattern)
//         2 = relu AFTER addend-add    ("relu(f(x) + g(x))")
// One wave per 16x16 C tile; M,N multiples of 16, K multiple of 32.
// Branch-free inner loop: pointer-incremented tiles + unconditional speculative
// prefetch (gfx1250 global_prefetch_b8; OOB translation failures drop silently).
// ---------------------------------------------------------------------------
__global__ __launch_bounds__(256)
void gemm_bf16_kernel(const float* __restrict__ A, int lda,
                      const float* __restrict__ Bm, int ldb,
                      const float* __restrict__ bias,
                      const float* __restrict__ addend, int ldadd,
                      float* __restrict__ C, int ldc,
                      int M, int Nn, int K, int relu) {
    int wid = (int)((blockIdx.x * blockDim.x + threadIdx.x) >> 5);
    int tilesN = Nn >> 4;
    int total = (M >> 4) * tilesN;
    if (wid >= total) return;                 // uniform per wave: EXEC stays all-ones
    int tm = wid / tilesN, tn = wid % tilesN;

    const float* Ap = A + (size_t)tm * 16 * lda;
    const float* Bp = Bm + tn * 16;
    const size_t bstep = (size_t)32 * ldb;
    v8f acc = {};
    for (int kk = 0; kk < K; kk += 32) {
        __builtin_prefetch(Ap + 32, 0, 1);    // next K-step (speculative)
        __builtin_prefetch(Bp + bstep, 0, 1);
        v16bf a = load_A_f32(Ap, lda);
        v16bf b = load_B_kn(Bp, ldb);
        acc = wmma_bf16(a, b, acc);
        Ap += 32;
        Bp += bstep;
    }
    int lane = threadIdx.x & 31;
    int n  = tn * 16 + (lane & 15);
    int m0 = tm * 16 + ((lane >> 4) << 3);
    float bv = bias ? bias[n] : 0.f;
#pragma unroll
    for (int r = 0; r < 8; ++r) {
        int m = m0 + r;
        float v = acc[r] + bv;
        if (relu == 1) v = fmaxf(v, 0.f);
        if (addend) v += addend[(size_t)m * ldadd + n];
        if (relu == 2) v = fmaxf(v, 0.f);
        C[(size_t)m * ldc + n] = v;
    }
}

// ---------------------------------------------------------------------------
// Node encoder: h0[n] = [emb_w[type], float(ninv)]
// ---------------------------------------------------------------------------
__global__ void encode_kernel(const int* __restrict__ x, const float* __restrict__ emb_w,
                              float* __restrict__ h0, int Nn) {
    int n = blockIdx.x * blockDim.x + threadIdx.x;
    if (n >= Nn) return;
    int t = x[2 * n], inv = x[2 * n + 1];
    h0[4 * n + 0] = emb_w[3 * t + 0];
    h0[4 * n + 1] = emb_w[3 * t + 1];
    h0[4 * n + 2] = emb_w[3 * t + 2];
    h0[4 * n + 3] = (float)inv;
}

__global__ void fill_u32_kernel(unsigned* __restrict__ p, unsigned v, int n) {
    int i = blockIdx.x * blockDim.x + threadIdx.x;
    if (i < n) p[i] = v;
}

// ---------------------------------------------------------------------------
// Segment softmax aggregation (per channel), 3 passes with atomics
// ---------------------------------------------------------------------------
__global__ void seg_max_kernel(const float* __restrict__ h, const int* __restrict__ ei,
                               const float* __restrict__ tptr, unsigned* __restrict__ mx,
                               int E_, int C) {
    int i = blockIdx.x * blockDim.x + threadIdx.x;
    int tot = E_ * C;
    if (i >= tot) return;
    int e = i / C, c = i - e * C;
    int s = ei[e], d = ei[E_ + e];
    float lg = h[(size_t)s * C + c] * tptr[0];
    atomicMax(&mx[(size_t)d * C + c], enc_f(lg));
}

__global__ void seg_sum_kernel(const float* __restrict__ h, const int* __restrict__ ei,
                               const float* __restrict__ tptr, const unsigned* __restrict__ mx,
                               float* __restrict__ num, float* __restrict__ den,
                               int E_, int C) {
    int i = blockIdx.x * blockDim.x + threadIdx.x;
    int tot = E_ * C;
    if (i >= tot) return;
    int e = i / C, c = i - e * C;
    int s = ei[e], d = ei[E_ + e];
    float v  = h[(size_t)s * C + c];
    float ex = __expf(v * tptr[0] - dec_f(mx[(size_t)d * C + c]));
    atomicAdd(&den[(size_t)d * C + c], ex);
    atomicAdd(&num[(size_t)d * C + c], v * ex);
}

__global__ void seg_div_kernel(float* __restrict__ num, const float* __restrict__ den, int tot) {
    int i = blockIdx.x * blockDim.x + threadIdx.x;
    if (i >= tot) return;
    float d = den[i];
    num[i] = (d > 0.f) ? num[i] / d : 0.f;
}

// conv1 linear (K=4, too small for WMMA): out = relu(ag@lW + lb + h0@rW)
__global__ void sage1_kernel(const float* __restrict__ ag, const float* __restrict__ h0,
                             const float* __restrict__ lW, const float* __restrict__ lb,
                             const float* __restrict__ rW, float* __restrict__ out, int Nn) {
    int i = blockIdx.x * blockDim.x + threadIdx.x;
    if (i >= Nn * 256) return;
    int n = i >> 8, j = i & 255;
    float s = lb[j];
#pragma unroll
    for (int k = 0; k < 4; ++k)
        s += ag[4 * n + k] * lW[k * 256 + j] + h0[4 * n + k] * rW[k * 256 + j];
    out[i] = fmaxf(s, 0.f);
}

// ---------------------------------------------------------------------------
// Set-transformer SAB attention, per (graph, 16-row Q tile): one wave.
// S = softmax(Q K^T / 16) in 32KB LDS, O = S @ V.  All matmuls via WMMA.
// ---------------------------------------------------------------------------
__global__ __launch_bounds__(32)
void attn_kernel(const float* __restrict__ Q, const float* __restrict__ Kb,
                 const float* __restrict__ V, float* __restrict__ O) {
    __shared__ float S[16 * 512];
    int b = blockIdx.x >> 5, qt = blockIdx.x & 31;
    int lane = threadIdx.x & 31;
    const float* Qt = Q + ((size_t)b * 512 + qt * 16) * 256;

    for (int kt = 0; kt < 32; ++kt) {
        v8f acc = {};
        const float* Kt = Kb + ((size_t)b * 512 + kt * 16) * 256;
        for (int kc = 0; kc < 256; kc += 32)
            acc = wmma_bf16(load_A_f32(Qt + kc, 256), load_A_f32(Kt + kc, 256), acc);
        int n = lane & 15, m0 = (lane >> 4) << 3;
#pragma unroll
        for (int r = 0; r < 8; ++r) S[(m0 + r) * 512 + kt * 16 + n] = acc[r] * 0.0625f;
    }
    __syncthreads();
    if (lane < 16) {
        float* row = &S[lane * 512];
        float m = -1e30f;
        for (int k = 0; k < 512; ++k) m = fmaxf(m, row[k]);
        float s = 0.f;
        for (int k = 0; k < 512; ++k) { float e = __expf(row[k] - m); row[k] = e; s += e; }
        float inv = 1.f / s;
        for (int k = 0; k < 512; ++k) row[k] *= inv;
    }
    __syncthreads();
    float* Or = O + ((size_t)b * 512 + qt * 16) * 256;
    for (int ct = 0; ct < 16; ++ct) {
        v8f acc = {};
        for (int k2 = 0; k2 < 512; k2 += 32) {
            v16bf a  = load_A_f32(&S[k2], 512);
            v16bf bb = load_B_kn(V + ((size_t)b * 512 + k2) * 256 + ct * 16, 256);
            acc = wmma_bf16(a, bb, acc);
        }
        int n = lane & 15, m0 = (lane >> 4) << 3;
#pragma unroll
        for (int r = 0; r < 8; ++r) Or[(m0 + r) * 256 + ct * 16 + n] = acc[r];
    }
}

// PMA: single-seed attention per graph (scores[512] -> softmax -> weighted V sum)
__global__ __launch_bounds__(256)
void pma_kernel(const float* __restrict__ q2, const float* __restrict__ K2,
                const float* __restrict__ V2, float* __restrict__ outp) {
    __shared__ float sc[512];
    __shared__ float red[2];
    int b = blockIdx.x, tid = threadIdx.x;
    for (int k = tid; k < 512; k += 256) {
        const float* kr = K2 + ((size_t)b * 512 + k) * 256;
        float s = 0.f;
        for (int h = 0; h < 256; ++h) s += q2[h] * kr[h];
        sc[k] = s * 0.0625f;
    }
    __syncthreads();
    if (tid == 0) {
        float m = -1e30f;
        for (int k = 0; k < 512; ++k) m = fmaxf(m, sc[k]);
        float s = 0.f;
        for (int k = 0; k < 512; ++k) s += __expf(sc[k] - m);
        red[0] = m; red[1] = s;
    }
    __syncthreads();
    float m = red[0], inv = 1.f / red[1];
    float acc = 0.f;
    for (int k = 0; k < 512; ++k)
        acc += __expf(sc[k] - m) * inv * V2[((size_t)b * 512 + k) * 256 + tid];
    outp[b * 256 + tid] = acc;
}

__global__ void vecmat_kernel(const float* __restrict__ v, const float* __restrict__ W,
                              const float* __restrict__ bias, float* __restrict__ o,
                              int K, int Nn) {
    int j = blockIdx.x * blockDim.x + threadIdx.x;
    if (j >= Nn) return;
    float s = bias[j];
    for (int k = 0; k < K; ++k) s += v[k] * W[k * Nn + j];
    o[j] = s;
}

__global__ void cvt_bf16_kernel(const float* __restrict__ s, __bf16* __restrict__ d, int n) {
    int i = blockIdx.x * blockDim.x + threadIdx.x;
    if (i < n) d[i] = (__bf16)s[i];
}

// ---------------------------------------------------------------------------
// GRU recurrence: persistent workgroup owns 16 batch rows, h lives in LDS,
// per-step gi = x_t@Wih^T, gh = h@Whh^T via WMMA (bf16 weights from L2),
// gate math entirely in registers.  blockDim=256 (8 waves);
// wave w owns output columns [32w, 32w+32) of all three gates.
// ---------------------------------------------------------------------------
__global__ __launch_bounds__(256)
void gru_kernel(const float* __restrict__ xd,          // [B,512,256]
                const __bf16* __restrict__ Wih,        // [768,256]
                const __bf16* __restrict__ Whh,        // [768,256]
                const float* __restrict__ bih, const float* __restrict__ bhh,
                float* __restrict__ outp, int ldo) {
    __shared__ float hsh[16 * 256];
    int wave = threadIdx.x >> 5, lane = threadIdx.x & 31;
    int b0 = blockIdx.x * 16;
    for (int i = threadIdx.x; i < 16 * 256; i += 256) hsh[i] = 0.f;
    __syncthreads();
    int colBase = wave * 32;

    for (int t = 0; t < 512; ++t) {
        v8f gir[2], giz[2], gin_[2], ghr[2], ghz[2], ghn[2];
#pragma unroll
        for (int s = 0; s < 2; ++s) { gir[s] = {}; giz[s] = {}; gin_[s] = {}; ghr[s] = {}; ghz[s] = {}; ghn[s] = {}; }

        for (int kk = 0; kk < 256; kk += 32) {
            v16bf ax = load_A_f32(xd + (size_t)b0 * 131072 + (size_t)t * 256 + kk, 131072);
            v16bf ah = load_A_f32(&hsh[kk], 256);
#pragma unroll
            for (int s = 0; s < 2; ++s) {
                int cb = colBase + s * 16;
                gir[s]  = wmma_bf16(ax, load_A_bf16(Wih + (size_t)(cb)       * 256 + kk, 256), gir[s]);
                giz[s]  = wmma_bf16(ax, load_A_bf16(Wih + (size_t)(256 + cb) * 256 + kk, 256), giz[s]);
                gin_[s] = wmma_bf16(ax, load_A_bf16(Wih + (size_t)(512 + cb) * 256 + kk, 256), gin_[s]);
                ghr[s]  = wmma_bf16(ah, load_A_bf16(Whh + (size_t)(cb)       * 256 + kk, 256), ghr[s]);
                ghz[s]  = wmma_bf16(ah, load_A_bf16(Whh + (size_t)(256 + cb) * 256 + kk, 256), ghz[s]);
                ghn[s]  = wmma_bf16(ah, load_A_bf16(Whh + (size_t)(512 + cb) * 256 + kk, 256), ghn[s]);
            }
        }
        __syncthreads();  // all waves done reading hsh
#pragma unroll
        for (int s = 0; s < 2; ++s) {
            int j = colBase + s * 16 + (lane & 15);
            float br = bih[j], bz = bih[256 + j], bn = bih[512 + j];
            float cr = bhh[j], cz = bhh[256 + j], cn = bhh[512 + j];
            int m0 = (lane >> 4) << 3;
#pragma unroll
            for (int r = 0; r < 8; ++r) {
                int m = m0 + r;
                float hold = hsh[m * 256 + j];
                float rg = sigmf(gir[s][r] + br + ghr[s][r] + cr);
                float zg = sigmf(giz[s][r] + bz + ghz[s][r] + cz);
                float ng = tanhf(gin_[s][r] + bn + rg * (ghn[s][r] + cn));
                hsh[m * 256 + j] = (1.f - zg) * ng + zg * hold;
            }
        }
        __syncthreads();
    }
    for (int i = threadIdx.x; i < 16 * 256; i += 256) {
        int m = i >> 8, j = i & 255;
        outp[(size_t)(b0 + m) * ldo + j] = hsh[i];
    }
}

// ---------------------------------------------------------------------------
// Orchestration
// ---------------------------------------------------------------------------
extern "C" void kernel_launch(void* const* d_in, const int* in_sizes, int n_in,
                              void* d_out, int out_size, void* d_ws, size_t ws_size,
                              hipStream_t stream) {
    const int N_ = 131072, E_ = 262144, H_ = 256, B_ = 256, T_ = 512, H3 = 768;
    const int HH = H_ * H_;

    const int*   x      = (const int*)d_in[0];
    const int*   ei     = (const int*)d_in[1];
    const float* emb_w  = (const float*)d_in[3];
    const float* c1_lW  = (const float*)d_in[4];
    const float* c1_lb  = (const float*)d_in[5];
    const float* c1_rW  = (const float*)d_in[6];
    const float* t1     = (const float*)d_in[7];
    const float* c2_lW  = (const float*)d_in[8];
    const float* c2_lb  = (const float*)d_in[9];
    const float* c2_rW  = (const float*)d_in[10];
    const float* t2     = (const float*)d_in[11];
    const float* mlp_W  = (const float*)d_in[12];
    const float* mlp_b  = (const float*)d_in[13];
    const float* gWih   = (const float*)d_in[14];
    const float* gWhh   = (const float*)d_in[15];
    const float* gbih   = (const float*)d_in[16];
    const float* gbhh   = (const float*)d_in[17];
    const float* st_Wq  = (const float*)d_in[18];
    const float* st_bq  = (const float*)d_in[19];
    const float* st_Wk  = (const float*)d_in[20];
    const float* st_bk  = (const float*)d_in[21];
    const float* st_Wv  = (const float*)d_in[22];
    const float* st_bv  = (const float*)d_in[23];
    const float* st_Wo  = (const float*)d_in[24];
    const float* st_bo  = (const float*)d_in[25];
    const float* st_lW  = (const float*)d_in[26];
    const float* st_lb  = (const float*)d_in[27];
    const float* pma_lW = (const float*)d_in[28];
    const float* pma_lb = (const float*)d_in[29];
    const float* seed   = (const float*)d_in[30];
    const float* fin_W  = (const float*)d_in[31];
    const float* fin_b  = (const float*)d_in[32];
    float* out = (float*)d_out;

    // workspace layout (with lifetime-based reuse of the 5 big N*H buffers)
    size_t NH = (size_t)N_ * H_;
    float* bufA = (float*)d_ws;        // h1 -> Q -> hA -> K2
    float* bufB = bufA + NH;           // max2 -> tmp -> K -> enc
    float* bufC = bufB + NH;           // num2/ag2 -> V -> kk
    float* bufD = bufC + NH;           // den2 -> h2 (= xd, kept alive to the end)
    float* bufE = bufD + NH;           // attnO -> V2
    float* h0   = bufE + NH;
    unsigned* mx1 = (unsigned*)(h0 + (size_t)N_ * 4);
    float* num1 = (float*)(mx1 + (size_t)N_ * 4);
    float* den1 = num1 + (size_t)N_ * 4;
    __bf16* wWih = (__bf16*)(den1 + (size_t)N_ * 4);
    __bf16* wWhh = wWih + (size_t)H3 * H_;
    float* q2v    = (float*)(wWhh + (size_t)H3 * H_);
    float* pmaout = q2v + H_;
    float* sm1 = pmaout + (size_t)B_ * H_;
    float* sm2 = sm1 + (size_t)B_ * H_;
    float* sm3 = sm2 + (size_t)B_ * H_;
    float* sm4 = sm3 + (size_t)B_ * H_;
    float* pooled = sm4 + (size_t)B_ * H_;   // [B,768]

    auto gemm = [&](const float* A, int lda, const float* Bm, int ldb,
                    const float* bias, const float* addend, int ldadd,
                    float* C, int ldc, int M, int Nn, int K, int relu) {
        int tiles = (M / 16) * (Nn / 16);
        gemm_bf16_kernel<<<(tiles + 7) / 8, 256, 0, stream>>>(
            A, lda, Bm, ldb, bias, addend, ldadd, C, ldc, M, Nn, K, relu);
    };

    // 0) node encoder
    encode_kernel<<<(N_ + 255) / 256, 256, 0, stream>>>(x, emb_w, h0, N_);

    // 1) SAGE conv1 (C=4, scalar path: K too small for WMMA)
    {
        int tot = N_ * 4;
        fill_u32_kernel<<<(tot + 255) / 256, 256, 0, stream>>>(mx1, ENC_NEG_INF, tot);
        hipMemsetAsync(num1, 0, (size_t)tot * 4, stream);
        hipMemsetAsync(den1, 0, (size_t)tot * 4, stream);
        int ec = E_ * 4;
        seg_max_kernel<<<(ec + 255) / 256, 256, 0, stream>>>(h0, ei, t1, mx1, E_, 4);
        seg_sum_kernel<<<(ec + 255) / 256, 256, 0, stream>>>(h0, ei, t1, mx1, num1, den1, E_, 4);
        seg_div_kernel<<<(tot + 255) / 256, 256, 0, stream>>>(num1, den1, tot);
        sage1_kernel<<<(N_ * 256 + 255) / 256, 256, 0, stream>>>(num1, h0, c1_lW, c1_lb, c1_rW, bufA, N_);
    }

    // 2) SAGE conv2 (C=256): scatter-softmax aggregation, then two WMMA GEMMs
    {
        int tot = (int)NH;
        unsigned* mx2 = (unsigned*)bufB;
        fill_u32_kernel<<<(tot + 255) / 256, 256, 0, stream>>>(mx2, ENC_NEG_INF, tot);
        hipMemsetAsync(bufC, 0, NH * 4, stream);
        hipMemsetAsync(bufD, 0, NH * 4, stream);
        int ec = E_ * 256;
        seg_max_kernel<<<(ec + 255) / 256, 256, 0, stream>>>(bufA, ei, t2, mx2, E_, 256);
        seg_sum_kernel<<<(ec + 255) / 256, 256, 0, stream>>>(bufA, ei, t2, mx2, bufC, bufD, E_, 256);
        seg_div_kernel<<<(tot + 255) / 256, 256, 0, stream>>>(bufC, bufD, tot);   // ag2 -> bufC
        // tmp = h1 @ c2_rW
        gemm(bufA, H_, c2_rW, H_, nullptr, nullptr, 0, bufB, H_, N_, H_, H_, 0);
        // h2 = relu(ag2 @ c2_lW + c2_lb + tmp)   (relu AFTER addend: mode 2)
        gemm(bufC, H_, c2_lW, H_, c2_lb, bufB, H_, bufD, H_, N_, H_, H_, 2);
    }

    // 3) p1 = xd.reshape(B, T*H) @ mlp_W + mlp_b  -> pooled[:, 0:256]
    gemm(bufD, T_ * H_, mlp_W, H_, mlp_b, nullptr, 0, pooled, H3, B_, H_, T_ * H_, 0);

    // 4) GRU -> pooled[:, 256:512]
    cvt_bf16_kernel<<<(H3 * H_ + 255) / 256, 256, 0, stream>>>(gWih, wWih, H3 * H_);
    cvt_bf16_kernel<<<(H3 * H_ + 255) / 256, 256, 0, stream>>>(gWhh, wWhh, H3 * H_);
    gru_kernel<<<B_ / 16, 256, 0, stream>>>(bufD, wWih, wWhh, gbih, gbhh, pooled + H_, H3);

    // 5) Set transformer
    gemm(bufD, H_, st_Wq, H_, st_bq, nullptr, 0, bufA, H_, N_, H_, H_, 0);   // Q
    gemm(bufD, H_, st_Wk, H_, st_bk, nullptr, 0, bufB, H_, N_, H_, H_, 0);   // K
    gemm(bufD, H_, st_Wv, H_, st_bv, nullptr, 0, bufC, H_, N_, H_, H_, 0);   // V
    attn_kernel<<<B_ * 32, 32, 0, stream>>>(bufA, bufB, bufC, bufE);         // O
    gemm(bufE, H_, st_Wo, H_, st_bo, bufD, H_, bufA, H_, N_, H_, H_, 0);     // hA = xd + O@Wo+bo
    gemm(bufA, H_, st_lW, H_, st_lb, bufA, H_, bufB, H_, N_, H_, H_, 1);     // enc = hA + relu(hA@Wl+bl)
    gemm(bufB, H_, pma_lW, H_, pma_lb, nullptr, 0, bufC, H_, N_, H_, H_, 1); // kk = relu(enc@pma_lW+b)
    gemm(bufC, H_, st_Wk + HH, H_, st_bk + H_, nullptr, 0, bufA, H_, N_, H_, H_, 0); // K2
    gemm(bufC, H_, st_Wv + HH, H_, st_bv + H_, nullptr, 0, bufE, H_, N_, H_, H_, 0); // V2
    vecmat_kernel<<<1, 256, 0, stream>>>(seed, st_Wq + HH, st_bq + H_, q2v, H_, H_);  // Q2
    pma_kernel<<<B_, 256, 0, stream>>>(q2v, bufA, bufE, pmaout);                       // attn out [B,H]
    gemm(pmaout, H_, st_Wo + HH, H_, st_bo + H_, seed, 0, sm1, H_, B_, H_, H_, 0);     // s1 = seed + o@Wo+bo
    gemm(sm1, H_, st_lW + HH, H_, st_lb + H_, sm1, H_, sm2, H_, B_, H_, H_, 1);        // s2 = s1 + relu(..)
    gemm(sm2, H_, st_Wv + 2 * HH, H_, st_bv + 2 * H_, nullptr, 0, sm3, H_, B_, H_, H_, 0); // v3
    gemm(sm3, H_, st_Wo + 2 * HH, H_, st_bo + 2 * H_, sm2, H_, sm4, H_, B_, H_, H_, 0);    // h3 = s2 + v3@Wo+bo
    gemm(sm4, H_, st_lW + 2 * HH, H_, st_lb + 2 * H_, sm4, H_, pooled + 2 * H_, H3, B_, H_, H_, 1); // p3

    // 6) final linear: out = pooled @ fin_W + fin_b
    gemm(pooled, H3, fin_W, H3, fin_b, nullptr, 0, out, H3, B_, H3, H3, 0);
}
